// StudentGNN_6597069766804
// MI455X (gfx1250) — matrix operand order repaired and checked by reading, not compiled, not run on hardware
//
#include <hip/hip_runtime.h>
#include <hip/hip_bf16.h>
#include <math.h>

typedef __attribute__((ext_vector_type(16))) __bf16 v16bf;
typedef __attribute__((ext_vector_type(8)))  __bf16 v8bf;
typedef __attribute__((ext_vector_type(8)))  float  v8f;

// ---------------------------------------------------------------------------
// Degree / normalization
// ---------------------------------------------------------------------------
__global__ void deg_kernel(float* __restrict__ deg, const int* __restrict__ dst, int E) {
    int e = blockIdx.x * blockDim.x + threadIdx.x;
    if (e >= E) return;
    __hip_atomic_fetch_add(&deg[dst[e]], 1.0f, __ATOMIC_RELAXED, __HIP_MEMORY_SCOPE_AGENT);
}

__global__ void dis_kernel(float* __restrict__ dis, const float* __restrict__ deg, int N) {
    int i = blockIdx.x * blockDim.x + threadIdx.x;
    if (i >= N) return;
    dis[i] = rsqrtf(deg[i] + 1.0f);   // self-loop contributes +1
}

// ---------------------------------------------------------------------------
// bf16 packing
// ---------------------------------------------------------------------------
__global__ void pack_bf16(const float* __restrict__ in, __bf16* __restrict__ out, long long n) {
    long long i = (long long)blockIdx.x * blockDim.x + threadIdx.x;
    if (i >= n) return;
    out[i] = (__bf16)in[i];
}

// W is [K x FOUT] row-major; produce WT [NPAD x K] (column n contiguous over k), zero-pad n>=FOUT
template <int K, int FOUT, int NPAD>
__global__ void pack_wT(const float* __restrict__ W, __bf16* __restrict__ WT) {
    int idx = blockIdx.x * blockDim.x + threadIdx.x;
    if (idx >= NPAD * K) return;
    int n = idx / K;
    int k = idx - n * K;
    float v = (n < FOUT) ? W[k * FOUT + n] : 0.0f;
    WT[idx] = (__bf16)v;
}

// ---------------------------------------------------------------------------
// WMMA GEMM: C[M x FOUT] = A[M x K] * B[K x FOUT]
// A: bf16 row-major.  BT: bf16 [NPAD x K] (B column-major).  One wave per 16x16 tile.
// NT = NPAD/16 column tiles.  M must be a multiple of 16 (100000 = 6250*16).
// ---------------------------------------------------------------------------
template <int K, int FOUT, int NT>
__global__ void gemm_wmma(const __bf16* __restrict__ A, const __bf16* __restrict__ BT,
                          float* __restrict__ C, int numTiles) {
    int wave = (blockIdx.x * blockDim.x + threadIdx.x) >> 5;   // uniform within wave
    if (wave >= numTiles) return;                               // whole-wave exit (EXEC all-1 for WMMA)
    int lane = threadIdx.x & 31;
    int tn = wave % NT;
    int tm = wave / NT;
    int m    = lane & 15;
    int half = lane >> 4;

    const __bf16* arow = A  + (size_t)(tm * 16 + m) * K;  // A fragment: row = M
    const __bf16* bcol = BT + (size_t)(tn * 16 + m) * K;  // B fragment: col = N

    v8f c = {};
#pragma unroll
    for (int kk = 0; kk < K; kk += 32) {
        // A fragment, contiguous k order chosen here (B matches): lanes0-15 hold K [kk..kk+15];
        //                      lanes16-31 hold [kk+16..kk+31]
        v8bf lo = *(const v8bf*)(arow + kk + half * 16);
        v8bf hi = *(const v8bf*)(arow + kk + half * 16 + 8);
        v16bf a;
#pragma unroll
        for (int i = 0; i < 8; ++i) { a[i] = lo[i]; a[8 + i] = hi[i]; }
        // B fragment, same contiguous k order as A: lane holds 16 consecutive K of its column
        v16bf b = *(const v16bf*)(bcol + kk + half * 16);
        c = __builtin_amdgcn_wmma_f32_16x16x32_bf16(false, a, false, b, (short)0, c, false, false);
    }

    int col = tn * 16 + m;
    if (col < FOUT) {
        // C/D f32 layout: lanes0-15 -> rows v (N=lane); lanes16-31 -> rows 8+v
#pragma unroll
        for (int v = 0; v < 8; ++v) {
            C[(size_t)(tm * 16 + v + 8 * half) * FOUT + col] = c[v];
        }
    }
}

// ---------------------------------------------------------------------------
// Edge-parallel scatter aggregation: agg[dst] += dis[src]*dis[dst]*h[src]
// Wave32 covers 32 consecutive idx = e*F+f (one edge if F=64, may span two if F=40) -> 128B coalesced gather + atomics.
// ---------------------------------------------------------------------------
template <int F>
__global__ void edge_agg(const float* __restrict__ h, float* __restrict__ agg,
                         const int* __restrict__ src, const int* __restrict__ dst,
                         const float* __restrict__ dis, long long total) {
    long long idx = (long long)blockIdx.x * blockDim.x + threadIdx.x;
    if (idx >= total) return;
    int e = (int)(idx / F);
    int f = (int)(idx - (long long)e * F);
    int s = src[e];
    int d = dst[e];
    float norm = dis[s] * dis[d];
    float val  = norm * h[(size_t)s * F + f];
    __hip_atomic_fetch_add(&agg[(size_t)d * F + f], val,
                           __ATOMIC_RELAXED, __HIP_MEMORY_SCOPE_AGENT);
}

// ---------------------------------------------------------------------------
// Layer-1 epilogue: self-loop + bias + ReLU, emit bf16 for next GEMM
// ---------------------------------------------------------------------------
__global__ void selfloop_relu_bf16(const float* __restrict__ agg, const float* __restrict__ h,
                                   const float* __restrict__ dis, const float* __restrict__ b,
                                   __bf16* __restrict__ outb, long long total /* N*64 */) {
    long long idx = (long long)blockIdx.x * blockDim.x + threadIdx.x;
    if (idx >= total) return;
    int i = (int)(idx >> 6);
    int f = (int)(idx & 63);
    float v = agg[idx] + dis[i] * dis[i] * h[idx] + b[f];
    outb[idx] = (__bf16)fmaxf(v, 0.0f);
}

// ---------------------------------------------------------------------------
// Layer-2 epilogue + log_softmax, one wave32 per row of 40.
// Lane l handles cols l and (l+32 for l<8); wave shuffles do max/sum reductions.
// ---------------------------------------------------------------------------
__global__ void finalize_logsoftmax(const float* __restrict__ agg2, const float* __restrict__ h2,
                                    const float* __restrict__ dis, const float* __restrict__ b2,
                                    float* __restrict__ out, int N) {
    int row = (blockIdx.x * blockDim.x + threadIdx.x) >> 5;   // uniform within wave
    if (row >= N) return;
    int lane = threadIdx.x & 31;
    float dii  = dis[row] * dis[row];
    size_t base = (size_t)row * 40;

    float v0 = agg2[base + lane] + dii * h2[base + lane] + b2[lane];
    float v1 = -INFINITY;
    if (lane < 8)
        v1 = agg2[base + 32 + lane] + dii * h2[base + 32 + lane] + b2[32 + lane];

    float m = fmaxf(v0, v1);
#pragma unroll
    for (int o = 16; o >= 1; o >>= 1) m = fmaxf(m, __shfl_xor(m, o, 32));

    float s = expf(v0 - m) + ((lane < 8) ? expf(v1 - m) : 0.0f);
#pragma unroll
    for (int o = 16; o >= 1; o >>= 1) s += __shfl_xor(s, o, 32);

    float lse = m + logf(s);
    out[base + lane] = v0 - lse;
    if (lane < 8) out[base + 32 + lane] = v1 - lse;
}

// ---------------------------------------------------------------------------
// Launch
// ---------------------------------------------------------------------------
extern "C" void kernel_launch(void* const* d_in, const int* in_sizes, int n_in,
                              void* d_out, int out_size, void* d_ws, size_t ws_size,
                              hipStream_t stream) {
    const float* x   = (const float*)d_in[0];   // [N,128]
    const int*   ei  = (const int*)d_in[1];     // [2,E]
    const float* W1  = (const float*)d_in[2];   // [128,64]
    const float* b1  = (const float*)d_in[3];   // [64]
    const float* W2  = (const float*)d_in[4];   // [64,40]
    const float* b2  = (const float*)d_in[5];   // [40]
    float* out = (float*)d_out;                 // [N,40]

    const int N = in_sizes[0] / 128;            // 100000 (multiple of 16)
    const int E = in_sizes[1] / 2;              // 1600000
    const int* src = ei;
    const int* dst = ei + E;

    // Workspace carve-out (256B aligned)
    char* w = (char*)d_ws;
    size_t off = 0;
    auto carve = [&](size_t bytes) -> char* {
        char* p = w + off;
        off = (off + bytes + 255) & ~(size_t)255;
        return p;
    };
    float*  deg    = (float*)carve((size_t)N * 4);
    float*  dis    = (float*)carve((size_t)N * 4);
    __bf16* xb     = (__bf16*)carve((size_t)N * 128 * 2);
    __bf16* W1T    = (__bf16*)carve((size_t)64 * 128 * 2);
    __bf16* W2T    = (__bf16*)carve((size_t)48 * 64 * 2);
    float*  h1     = (float*)carve((size_t)N * 64 * 4);
    float*  agg1   = (float*)carve((size_t)N * 64 * 4);
    __bf16* hrelub = (__bf16*)carve((size_t)N * 64 * 2);
    float*  h2     = (float*)carve((size_t)N * 40 * 4);
    float*  agg2   = (float*)carve((size_t)N * 40 * 4);
    (void)ws_size;

    const int T = 256;
    auto blk = [](long long n, int t) { return (int)((n + t - 1) / t); };

    // --- degrees & normalization ---
    hipMemsetAsync(deg, 0, (size_t)N * 4, stream);
    deg_kernel<<<blk(E, T), T, 0, stream>>>(deg, dst, E);
    dis_kernel<<<blk(N, T), T, 0, stream>>>(dis, deg, N);

    // --- packing to bf16 ---
    pack_bf16<<<blk((long long)N * 128, T), T, 0, stream>>>(x, xb, (long long)N * 128);
    pack_wT<128, 64, 64><<<blk(64 * 128, T), T, 0, stream>>>(W1, W1T);
    pack_wT<64, 40, 48><<<blk(48 * 64, T), T, 0, stream>>>(W2, W2T);

    // --- layer 1: h1 = x @ W1 (bf16 WMMA, f32 accum) ---
    {
        int tiles = (N / 16) * 4;                       // 4 column tiles of 16 (64 out)
        gemm_wmma<128, 64, 4><<<blk((long long)tiles * 32, T), T, 0, stream>>>(xb, W1T, h1, tiles);
    }
    hipMemsetAsync(agg1, 0, (size_t)N * 64 * 4, stream);
    edge_agg<64><<<blk((long long)E * 64, T), T, 0, stream>>>(h1, agg1, src, dst, dis, (long long)E * 64);
    selfloop_relu_bf16<<<blk((long long)N * 64, T), T, 0, stream>>>(agg1, h1, dis, b1, hrelub, (long long)N * 64);

    // --- layer 2: h2 = relu(h) @ W2 ---
    {
        int tiles = (N / 16) * 3;                       // 3 column tiles (40 out, padded to 48)
        gemm_wmma<64, 40, 3><<<blk((long long)tiles * 32, T), T, 0, stream>>>(hrelub, W2T, h2, tiles);
    }
    hipMemsetAsync(agg2, 0, (size_t)N * 40 * 4, stream);
    edge_agg<40><<<blk((long long)E * 40, T), T, 0, stream>>>(h2, agg2, src, dst, dis, (long long)E * 40);

    // --- epilogue + log_softmax (wave per row) ---
    finalize_logsoftmax<<<blk((long long)N * 32, T), T, 0, stream>>>(agg2, h2, dis, b2, out, N);
}
